// LSTM_29712583754288
// MI455X (gfx1250) — matrix operand order repaired and checked by reading, not compiled, not run on hardware
//
#include <hip/hip_runtime.h>
#include <hip/hip_bf16.h>

typedef __attribute__((ext_vector_type(16))) __bf16 v16bf;
typedef __attribute__((ext_vector_type(8)))  __bf16 v8bf;
typedef __attribute__((ext_vector_type(8)))  float  v8f;

#define HID    1024
#define BATCH  128
#define DIN    512
#define SEQ    512
#define DOUT   512
#define NBLK   64      // persistent grid: 64 blocks x 256 threads = 512 waves

// ---- optimization launder ------------------------------------------------
// Offset-launder keeps pointer provenance (=> global_load, not flat_load)
// while making addresses non-loop-invariant (=> no LICM hoist/spill).
__device__ __forceinline__ const __bf16* opq_ptr(const __bf16* p) {
    long long z = 0;
    asm volatile("" : "+s"(z));
    return p + z;
}
__device__ __forceinline__ int opq_i32(int v) {
    asm volatile("" : "+s"(v));
    return v;
}

// ---------------------------------------------------------------------------
// WMMA fragment loads (CDNA5 16x16x32 bf16 layouts, wave32)
// A 16x32 (MxK): lane<16 -> row=lane,    K {0..7, 16..23}
//                lane>=16 -> row=lane-16, K {8..15, 24..31}
// B 32x16 (KxN): lane<16 -> col=lane, K {0..7, 16..23}; lane>=16 -> K {8..15, 24..31}
// C/D 16x16 f32: VGPR r, lane l -> M = r + 8*(l>=16), N = l&15
// ---------------------------------------------------------------------------
__device__ __forceinline__ v16bf load_a_frag(const __bf16* rowPtr, int k, int half) {
    union { v16bf v; v8bf h[2]; } u;
    u.h[0] = *(const v8bf*)(rowPtr + k + half * 8);
    u.h[1] = *(const v8bf*)(rowPtr + k + 16 + half * 8);
    return u.v;
}
__device__ __forceinline__ v16bf load_b_frag(const __bf16* rowPtr, int k, int half) {
    union { v16bf v; v8bf h[2]; } u;
    u.h[0] = *(const v8bf*)(rowPtr + k + half * 8);
    u.h[1] = *(const v8bf*)(rowPtr + k + 16 + half * 8);
    return u.v;
}
__device__ __forceinline__ v8f wmma_bf16(v16bf a, v16bf b, v8f c) {
    return __builtin_amdgcn_wmma_f32_16x16x32_bf16(false, a, false, b, (short)0, c, false, false);
}
__device__ __forceinline__ float sigmoidf_(float x) { return 1.0f / (1.0f + __expf(-x)); }
__device__ __forceinline__ float tanhf_(float x) {
    float e = __expf(-2.0f * x);
    return (1.0f - e) / (1.0f + e);
}

// Branch-free K-segment: 4 gate accumulators share one A stream.
// K arrives as an opaque runtime value -> loop stays rolled, unrolled x2.
__device__ __forceinline__ void gemm4_seg(const __bf16* __restrict__ arow,
                                          const __bf16* __restrict__ w0,
                                          const __bf16* __restrict__ w1,
                                          const __bf16* __restrict__ w2,
                                          const __bf16* __restrict__ w3,
                                          int K, int half,
                                          v8f& ai, v8f& af, v8f& ag, v8f& ao) {
    v16bf a  = load_a_frag(arow, 0, half);
    v16bf b0 = load_b_frag(w0,   0, half);
    v16bf b1 = load_b_frag(w1,   0, half);
    v16bf b2 = load_b_frag(w2,   0, half);
    v16bf b3 = load_b_frag(w3,   0, half);
    #pragma unroll 2
    for (int k = 32; k < K; k += 32) {
        v16bf an  = load_a_frag(arow, k, half);
        v16bf b0n = load_b_frag(w0,   k, half);
        v16bf b1n = load_b_frag(w1,   k, half);
        v16bf b2n = load_b_frag(w2,   k, half);
        v16bf b3n = load_b_frag(w3,   k, half);
        ai = wmma_bf16(a, b0, ai);
        af = wmma_bf16(a, b1, af);
        ag = wmma_bf16(a, b2, ag);
        ao = wmma_bf16(a, b3, ao);
        a = an; b0 = b0n; b1 = b1n; b2 = b2n; b3 = b3n;
    }
    ai = wmma_bf16(a, b0, ai);
    af = wmma_bf16(a, b1, af);
    ag = wmma_bf16(a, b2, ag);
    ao = wmma_bf16(a, b3, ao);
}

// One 16(batch) x 16(hidden) patch of a fused LSTM cell.
__device__ __forceinline__ void cell_tile(const __bf16* __restrict__ A0, int lda0,
                                          const __bf16* __restrict__ A1,
                                          int K1, int Ktot,
                                          const __bf16* __restrict__ W,
                                          const float* __restrict__ bias,
                                          float* __restrict__ c,
                                          __bf16* __restrict__ h_out,
                                          int m0, int n0, int half, int lr) {
    const __bf16* a0row = A0 + (size_t)(m0 + lr) * lda0;
    const __bf16* a1row = A1 + (size_t)(m0 + lr) * HID;
    const __bf16* wrow0 = W + (size_t)(0 * HID + n0 + lr) * Ktot;
    const __bf16* wrow1 = W + (size_t)(1 * HID + n0 + lr) * Ktot;
    const __bf16* wrow2 = W + (size_t)(2 * HID + n0 + lr) * Ktot;
    const __bf16* wrow3 = W + (size_t)(3 * HID + n0 + lr) * Ktot;

    v8f ai = {}, af = {}, ag = {}, ao = {};
    gemm4_seg(a0row, wrow0, wrow1, wrow2, wrow3, K1, half, ai, af, ag, ao);
    gemm4_seg(a1row, wrow0 + K1, wrow1 + K1, wrow2 + K1, wrow3 + K1,
              Ktot - K1, half, ai, af, ag, ao);

    const int col = n0 + lr;
    const float bi_ = bias[0 * HID + col];
    const float bf_ = bias[1 * HID + col];
    const float bg_ = bias[2 * HID + col];
    const float bo_ = bias[3 * HID + col];

    #pragma unroll
    for (int r = 0; r < 8; ++r) {
        const int row   = m0 + r + half * 8;
        const size_t ix = (size_t)row * HID + col;
        const float iv = sigmoidf_(ai[r] + bi_);
        const float fv = sigmoidf_(af[r] + bf_);
        const float gv = tanhf_(ag[r] + bg_);
        const float ov = sigmoidf_(ao[r] + bo_);
        const float cn = fv * c[ix] + iv * gv;
        c[ix]     = cn;
        h_out[ix] = (__bf16)(ov * tanhf_(cn));
    }
}

// Device-wide barrier: monotonic target, release fence on entry, acquire spin.
__device__ __forceinline__ void grid_sync(unsigned int* __restrict__ bar,
                                          unsigned int target) {
    __threadfence();                 // release h/c stores device-wide (L2)
    __syncthreads();
    if (threadIdx.x == 0) {
        __hip_atomic_fetch_add(bar, 1u, __ATOMIC_ACQ_REL, __HIP_MEMORY_SCOPE_AGENT);
        while (__hip_atomic_load(bar, __ATOMIC_ACQUIRE, __HIP_MEMORY_SCOPE_AGENT) < target)
            __builtin_amdgcn_s_sleep(1);
    }
    __syncthreads();
}

// ---------------------------------------------------------------------------
// Persistent 2-layer LSTM: the whole recurrence in ONE kernel.
// 512 waves; wave -> (m_tile = wave>>6, n_tile = wave&63) gate patch.
// Per step: layer1 tiles -> grid_sync -> layer2 tiles -> grid_sync.
// Weight addresses are offset-laundered per timestep: loads stay in the step
// body (streamed from L2 as global_load_b128) and cannot be hoisted/spilled.
// ---------------------------------------------------------------------------
__global__ __launch_bounds__(256, 1)
void lstm_persistent_kernel(const __bf16* __restrict__ x_bf,
                            const __bf16* __restrict__ Wcat1,
                            const __bf16* __restrict__ Wcat2,
                            const __bf16* __restrict__ Wo,
                            const float* __restrict__ bias1,
                            const float* __restrict__ bias2,
                            float* __restrict__ c1,
                            float* __restrict__ c2,
                            __bf16* __restrict__ h1b0, __bf16* __restrict__ h1b1,
                            __bf16* __restrict__ h2b0, __bf16* __restrict__ h2b1,
                            const float* __restrict__ b_out,
                            float* __restrict__ out,
                            unsigned int* __restrict__ bar)
{
    const int wave   = (int)((blockIdx.x * blockDim.x + threadIdx.x) >> 5);
    const int lane   = (int)(threadIdx.x & 31);
    const int n_tile = wave & 63;   // HID/16
    const int m_tile = wave >> 6;   // BATCH/16
    const int half   = lane >> 4;
    const int lr     = lane & 15;
    const int m0     = m_tile * 16;
    const int n0     = n_tile * 16;

    // opaque trip counts: keep the K-loops rolled (no full unroll -> no spill)
    const int kDIN = opq_i32(DIN);
    const int kHID = opq_i32(HID);
    const int kT1  = opq_i32(DIN + HID);
    const int kT2  = opq_i32(2 * HID);

    unsigned int phase = 0;

    #pragma unroll 1
    for (int t = 0; t < SEQ; ++t) {
        const __bf16* h1_prev = (t & 1) ? h1b1 : h1b0;
        __bf16*       h1_new  = (t & 1) ? h1b0 : h1b1;
        const __bf16* h2_prev = (t & 1) ? h2b1 : h2b0;
        __bf16*       h2_new  = (t & 1) ? h2b0 : h2b1;

        // re-laundered every iteration: weight loads cannot be hoisted out,
        // provenance preserved -> global_load path
        const __bf16* W1 = opq_ptr(Wcat1);
        const __bf16* W2 = opq_ptr(Wcat2);

        // layer 1: gates = [x_t | h1] x Wcat1^T,  K = 512 + 1024
        cell_tile(x_bf + (size_t)t * DIN, SEQ * DIN, h1_prev,
                  kDIN, kT1, W1, bias1, c1, h1_new, m0, n0, half, lr);
        grid_sync(bar, (++phase) * NBLK);

        // layer 2: gates = [h1_new | h2] x Wcat2^T, K = 1024 + 1024
        cell_tile(h1_new, HID, h2_prev,
                  kHID, kT2, W2, bias2, c2, h2_new, m0, n0, half, lr);
        grid_sync(bar, (++phase) * NBLK);
    }

    // ---- final projection: out = h2 @ w_out^T + b_out  (8 x 32 tiles)
    if (wave < 256) {
        const __bf16* h2f  = (SEQ & 1) ? h2b1 : h2b0;
        const int on_tile  = wave & 31;   // DOUT/16
        const int om_tile  = wave >> 5;   // BATCH/16
        const int om0 = om_tile * 16, on0 = on_tile * 16;

        const __bf16* arow = h2f + (size_t)(om0 + lr) * HID;
        const __bf16* wrow = opq_ptr(Wo) + (size_t)(on0 + lr) * HID;

        v8f acc = {};
        v16bf a = load_a_frag(arow, 0, half);
        v16bf b = load_b_frag(wrow, 0, half);
        #pragma unroll 2
        for (int k = 32; k < kHID; k += 32) {
            v16bf an = load_a_frag(arow, k, half);
            v16bf bn = load_b_frag(wrow, k, half);
            acc = wmma_bf16(a, b, acc);
            a = an; b = bn;
        }
        acc = wmma_bf16(a, b, acc);

        const int col = on0 + lr;
        const float bo_ = b_out[col];
        #pragma unroll
        for (int r = 0; r < 8; ++r) {
            const int row = om0 + r + half * 8;
            out[(size_t)row * DOUT + col] = acc[r] + bo_;
        }
    }
}

// ---------------------------- prep kernels ---------------------------------
__global__ void cvt_f32_bf16_kernel(const float* __restrict__ in, __bf16* __restrict__ out, int n) {
    for (int i = blockIdx.x * blockDim.x + threadIdx.x; i < n; i += gridDim.x * blockDim.x)
        out[i] = (__bf16)in[i];
}

__global__ void pack_w_kernel(const float* __restrict__ w_ih, int Kin,
                              const float* __restrict__ w_hh, int Kh,
                              __bf16* __restrict__ out, int Ktot) {
    const int total = 4 * HID * Ktot;
    for (int i = blockIdx.x * blockDim.x + threadIdx.x; i < total; i += gridDim.x * blockDim.x) {
        const int n = i / Ktot, k = i % Ktot;
        const float v = (k < Kin) ? w_ih[(size_t)n * Kin + k]
                                  : w_hh[(size_t)n * Kh + (k - Kin)];
        out[i] = (__bf16)v;
    }
}

__global__ void bias_sum_kernel(const float* __restrict__ a, const float* __restrict__ b,
                                float* __restrict__ out, int n) {
    for (int i = blockIdx.x * blockDim.x + threadIdx.x; i < n; i += gridDim.x * blockDim.x)
        out[i] = a[i] + b[i];
}

__global__ void zero_kernel(unsigned int* __restrict__ p, int n) {
    for (int i = blockIdx.x * blockDim.x + threadIdx.x; i < n; i += gridDim.x * blockDim.x)
        p[i] = 0u;
}

// ---------------------------------------------------------------------------
extern "C" void kernel_launch(void* const* d_in, const int* in_sizes, int n_in,
                              void* d_out, int out_size, void* d_ws, size_t ws_size,
                              hipStream_t stream) {
    const float* x     = (const float*)d_in[0];
    const float* w_ih1 = (const float*)d_in[1];
    const float* w_hh1 = (const float*)d_in[2];
    const float* b_ih1 = (const float*)d_in[3];
    const float* b_hh1 = (const float*)d_in[4];
    const float* w_ih2 = (const float*)d_in[5];
    const float* w_hh2 = (const float*)d_in[6];
    const float* b_ih2 = (const float*)d_in[7];
    const float* b_hh2 = (const float*)d_in[8];
    const float* w_out = (const float*)d_in[9];
    const float* b_out = (const float*)d_in[10];

    char* p = (char*)d_ws;
    __bf16* x_bf  = (__bf16*)p;  p += (size_t)BATCH * SEQ * DIN * 2;       // 64 MiB
    __bf16* Wcat1 = (__bf16*)p;  p += (size_t)4 * HID * (DIN + HID) * 2;   // 12 MiB
    __bf16* Wcat2 = (__bf16*)p;  p += (size_t)4 * HID * (2 * HID) * 2;     // 16 MiB
    __bf16* Wo_bf = (__bf16*)p;  p += (size_t)DOUT * HID * 2;              // 1 MiB
    float*  bias1 = (float*)p;   p += (size_t)4 * HID * 4;
    float*  bias2 = (float*)p;   p += (size_t)4 * HID * 4;
    unsigned int* state = (unsigned int*)p;          // zeroed block starts here
    float*  c1    = (float*)p;   p += (size_t)BATCH * HID * 4;
    float*  c2    = (float*)p;   p += (size_t)BATCH * HID * 4;
    __bf16* h1b0  = (__bf16*)p;  p += (size_t)BATCH * HID * 2;
    __bf16* h1b1  = (__bf16*)p;  p += (size_t)BATCH * HID * 2;
    __bf16* h2b0  = (__bf16*)p;  p += (size_t)BATCH * HID * 2;
    __bf16* h2b1  = (__bf16*)p;  p += (size_t)BATCH * HID * 2;
    unsigned int* bar = (unsigned int*)p; p += 256;  // barrier counter (zeroed)
    const int state_words = (BATCH * HID * 16 + 256) / 4;

    // ---- prep: bf16 conversions, packed weights, fused biases, zero state+bar
    cvt_f32_bf16_kernel<<<2048, 256, 0, stream>>>(x, x_bf, BATCH * SEQ * DIN);
    pack_w_kernel<<<2048, 256, 0, stream>>>(w_ih1, DIN, w_hh1, HID, Wcat1, DIN + HID);
    pack_w_kernel<<<2048, 256, 0, stream>>>(w_ih2, HID, w_hh2, HID, Wcat2, 2 * HID);
    cvt_f32_bf16_kernel<<<512, 256, 0, stream>>>(w_out, Wo_bf, DOUT * HID);
    bias_sum_kernel<<<16, 256, 0, stream>>>(b_ih1, b_hh1, bias1, 4 * HID);
    bias_sum_kernel<<<16, 256, 0, stream>>>(b_ih2, b_hh2, bias2, 4 * HID);
    zero_kernel<<<512, 256, 0, stream>>>(state, state_words);

    // ---- the entire recurrence + projection: one persistent kernel
    lstm_persistent_kernel<<<NBLK, 256, 0, stream>>>(
        x_bf, Wcat1, Wcat2, Wo_bf, bias1, bias2,
        c1, c2, h1b0, h1b1, h2b0, h2b1, b_out, (float*)d_out, bar);
}
